// NonLocalBlockND_32392643346905
// MI455X (gfx1250) — matrix operand order repaired and checked by reading, not compiled, not run on hardware
//
#include <hip/hip_runtime.h>
#include <hip/hip_bf16.h>

typedef __attribute__((ext_vector_type(16))) _Float16 v16h;
typedef __attribute__((ext_vector_type(8)))  float    v8f;
typedef __attribute__((ext_vector_type(8)))  _Float16 h8;

#define NPIX   32768     // B*H*W
#define C_DIM  256
#define CI_DIM 128
#define NS     2048      // H*(W/2)
#define NQ     4096      // H*W per batch
#define BATCH  8

__device__ __forceinline__ v8f zero8() {
    v8f z;
#pragma unroll
    for (int i = 0; i < 8; ++i) z[i] = 0.f;
    return z;
}

__device__ __forceinline__ v8f wmma16(v16h a, v16h b, v8f c) {
    return __builtin_amdgcn_wmma_f32_16x16x32_f16(
        false, a, false, b, (short)0, c, false, false);
}

// ---------------- CDNA5 async global->LDS (ASYNCcnt-tracked, no VGPR bounce) ----

__device__ __forceinline__ void async_b128(void* lds, const void* g) {
    unsigned l = (unsigned)(size_t)(__attribute__((address_space(3))) void*)lds;
    unsigned long long a = (unsigned long long)g;
    asm volatile("global_load_async_to_lds_b128 %0, %1, off"
                 :: "v"(l), "v"(a) : "memory");
}

__device__ __forceinline__ void wait_async0() {
    asm volatile("s_wait_asynccnt 0" ::: "memory");
}

// cooperative async global->LDS tile copy, 16B per lane per op; cols multiple
// of 8, dld multiple of 8 halves so every row stays 16B aligned.
__device__ __forceinline__ void g2s_async(_Float16* dst, int dld,
                                          const _Float16* __restrict__ src, int sld,
                                          int rows, int cols, int tid, int nthr) {
    const int cw = cols >> 3;
    for (int i = tid; i < rows * cw; i += nthr) {
        int r  = i / cw;
        int cc = (i - r * cw) << 3;
        async_b128(dst + r * dld + cc, src + (size_t)r * sld + cc);
    }
}

// ---------------- WMMA fragment loaders (ISA 7.12.2 layouts) ----------------

// A fragment (16 rows x 32 K) from row-major [row][k], leading dim ld (halves).
__device__ __forceinline__ v16h loadA(const _Float16* p, int ld, int lane) {
    int row = lane & 15;
    int kb  = (lane >> 4) << 3;
    const _Float16* q = p + row * ld + kb;
    h8 lo = *(const h8*)(q);
    h8 hi = *(const h8*)(q + 16);
    return __builtin_shufflevector(lo, hi, 0,1,2,3,4,5,6,7,8,9,10,11,12,13,14,15);
}

// B fragment (32 K x 16 cols) where the source is stored [n][k]-major:
// B[k][col] = src[col][k]; per lane k = 8h..8h+7 and 16+8h..16+8h+7 -> two b128 loads.
__device__ __forceinline__ v16h loadB_nk(const _Float16* p, int ld, int lane) {
    int col = lane & 15;
    int kb  = (lane >> 4) << 3;
    const _Float16* q = p + col * ld + kb;
    h8 lo = *(const h8*)(q);
    h8 hi = *(const h8*)(q + 16);
    return __builtin_shufflevector(lo, hi, 0,1,2,3,4,5,6,7,8,9,10,11,12,13,14,15);
}

// ---------------- elementwise kernels ----------------

// f32 [K][N] -> f16 [N][K] (transpose so GEMM B-fragments are contiguous)
__global__ void cvt_transpose(const float* __restrict__ src, _Float16* __restrict__ dst,
                              int K, int N) {
    int i = blockIdx.x * blockDim.x + threadIdx.x;
    if (i >= K * N) return;
    int k = i / N, n = i - k * N;
    dst[(size_t)n * K + k] = (_Float16)src[i];
}

__global__ void bn_elu_f16(const float* __restrict__ x, const float* __restrict__ gamma,
                           const float* __restrict__ beta, const float* __restrict__ mean,
                           const float* __restrict__ var, _Float16* __restrict__ out,
                           int cmask, int n) {
    int i = blockIdx.x * blockDim.x + threadIdx.x;
    if (i >= n) return;
    int c = i & cmask;
    float v = gamma[c] * (x[i] - mean[c]) * rsqrtf(var[c] + 1e-3f) + beta[c];
    v = v > 0.f ? v : expm1f(v);
    out[i] = (_Float16)v;
}

// width-2 max pool; phi kept [b][m][ci] (row-major), g written [b][ci][m]
// (channel-major) so the attention O-update B-fragments are contiguous.
__global__ void pool_w2(const _Float16* __restrict__ gf, const _Float16* __restrict__ pf,
                        _Float16* __restrict__ gpt, _Float16* __restrict__ pp, int n) {
    int i = blockIdx.x * blockDim.x + threadIdx.x;  // over B*NS*CI
    if (i >= n) return;
    int ci = i & 127;
    int m  = (i >> 7) & 2047;       // h*32 + wp
    int b  = i >> 18;
    int h = m >> 5, wp = m & 31;
    size_t nsrc = ((size_t)((b * 64 + h) * 64 + 2 * wp)) * CI_DIM + ci;
    _Float16 a0 = gf[nsrc], a1 = gf[nsrc + CI_DIM];
    gpt[((size_t)(b * CI_DIM + ci)) * NS + m] = a0 > a1 ? a0 : a1;
    _Float16 b0 = pf[nsrc], b1 = pf[nsrc + CI_DIM];
    pp[i] = b0 > b1 ? b0 : b1;
}

// ---------------- projection GEMM: [NPIX,256] x Bt[128,256] + bias -> f16 ----------------

__global__ void __launch_bounds__(128) proj_gemm(const _Float16* __restrict__ A,
                                                 const _Float16* __restrict__ Bt,
                                                 const float* __restrict__ bias,
                                                 _Float16* __restrict__ out) {
    __shared__ _Float16 As[64 * 264];    // A tile  [m][k=256]
    __shared__ _Float16 Bs[128 * 264];   // weightT [n=128][k=256]
    const int tid  = threadIdx.x;
    const int wave = tid >> 5;
    const int lane = tid & 31;
    const int m0   = blockIdx.x * 64;

    g2s_async(As, 264, A + (size_t)m0 * C_DIM, C_DIM, 64, 256, tid, 128);
    g2s_async(Bs, 264, Bt, C_DIM, 128, 256, tid, 128);
    wait_async0();
    __syncthreads();

    v8f acc[8];
#pragma unroll
    for (int n = 0; n < 8; ++n) acc[n] = zero8();

#pragma unroll
    for (int kc = 0; kc < 8; ++kc) {     // K = 256 in 32-chunks
        v16h a = loadA(As + (wave * 16) * 264 + kc * 32, 264, lane);
#pragma unroll
        for (int n = 0; n < 8; ++n) {
            v16h b = loadB_nk(Bs + (n * 16) * 264 + kc * 32, 264, lane);
            acc[n] = wmma16(a, b, acc[n]);
        }
    }

    // epilogue: bias + f16 store (C/D layout: row=(lane>>4)*8+r, col=lane&15)
    const int rbase = m0 + wave * 16 + ((lane >> 4) << 3);
    const int cb    = lane & 15;
#pragma unroll
    for (int n = 0; n < 8; ++n) {
        int col  = n * 16 + cb;
        float bb = bias[col];
#pragma unroll
        for (int r = 0; r < 8; ++r)
            out[(size_t)(rbase + r) * CI_DIM + col] = (_Float16)(acc[n][r] + bb);
    }
}

// ---------------- fused flash attention: y = softmax(theta phi^T) g ----------------

__global__ void __launch_bounds__(128) attn_flash(const _Float16* __restrict__ theta,
                                                  const _Float16* __restrict__ phi,
                                                  const _Float16* __restrict__ gT,
                                                  float* __restrict__ y) {
    __shared__ _Float16 Ps[2][64 * 136];    // phi tile [m][k=128], double-buffered
    __shared__ _Float16 Gs[2][128 * 72];    // g tile   [ci=128][m=64], double-buffered
    __shared__ _Float16 Pw[4 * 16 * 72];    // per-wave P scratch (16x64)

    const int tid  = threadIdx.x;
    const int wave = tid >> 5;
    const int lane = tid & 31;
    const int b    = blockIdx.x >> 6;
    const int m0   = (blockIdx.x & 63) * 64;
    const int rg0  = b * NQ + m0 + wave * 16;   // global query row base of this wave

    // resident theta A fragments: 16 rows x 128 K  -> 4 fragments
    v16h ta[4];
    {
        int row = rg0 + (lane & 15);
        int kb  = (lane >> 4) << 3;
        const _Float16* tr = theta + (size_t)row * CI_DIM + kb;
#pragma unroll
        for (int kc = 0; kc < 4; ++kc) {
            h8 lo = *(const h8*)(tr + kc * 32);
            h8 hi = *(const h8*)(tr + kc * 32 + 16);
            ta[kc] = __builtin_shufflevector(lo, hi, 0,1,2,3,4,5,6,7,8,9,10,11,12,13,14,15);
        }
    }

    float rmax[8], rsum[8];
#pragma unroll
    for (int r = 0; r < 8; ++r) { rmax[r] = -1e30f; rsum[r] = 0.f; }
    v8f o[8];
#pragma unroll
    for (int c = 0; c < 8; ++c) o[c] = zero8();

    _Float16* pw = Pw + wave * 16 * 72;
    const _Float16* pbase = phi + (size_t)b * NS * CI_DIM;   // [m][k]
    const _Float16* gbase = gT  + (size_t)b * CI_DIM * NS;   // [ci][m]

    // prologue: async-stage tile 0 into buffer 0
    g2s_async(Ps[0], 136, pbase, CI_DIM, 64, 128, tid, 128);
    g2s_async(Gs[0], 72,  gbase, NS,     128, 64, tid, 128);
    wait_async0();
    __syncthreads();

    for (int mi = 0; mi < NS / 64; ++mi) {
        const int cur = mi & 1, nxt = cur ^ 1;
        // prefetch next tile while computing on the current one
        if (mi + 1 < NS / 64) {
            g2s_async(Ps[nxt], 136, pbase + (size_t)(mi + 1) * 64 * CI_DIM,
                      CI_DIM, 64, 128, tid, 128);
            g2s_async(Gs[nxt], 72,  gbase + (size_t)(mi + 1) * 64,
                      NS, 128, 64, tid, 128);
        }

        // S = theta_strip(16x128) * phi_tile^T(128x64) -> 16x64
        v8f s[4];
#pragma unroll
        for (int n = 0; n < 4; ++n) s[n] = zero8();
#pragma unroll
        for (int n = 0; n < 4; ++n)
#pragma unroll
            for (int kc = 0; kc < 4; ++kc) {
                v16h bf = loadB_nk(Ps[cur] + (n * 16) * 136 + kc * 32, 136, lane);
                s[n] = wmma16(ta[kc], bf, s[n]);
            }

        // online softmax (rows live across 16-lane halves of the wave)
        float scale[8];
#pragma unroll
        for (int r = 0; r < 8; ++r) {
            float mloc = fmaxf(fmaxf(s[0][r], s[1][r]), fmaxf(s[2][r], s[3][r]));
#pragma unroll
            for (int off = 1; off < 16; off <<= 1)
                mloc = fmaxf(mloc, __shfl_xor(mloc, off, 32));
            float mnew = fmaxf(rmax[r], mloc);
            scale[r] = __expf(rmax[r] - mnew);
            rmax[r]  = mnew;
            float psum = 0.f;
            int prow = ((lane >> 4) << 3) + r;
#pragma unroll
            for (int n = 0; n < 4; ++n) {
                float pv = __expf(s[n][r] - mnew);
                psum += pv;
                pw[prow * 72 + n * 16 + (lane & 15)] = (_Float16)pv;
            }
#pragma unroll
            for (int off = 1; off < 16; off <<= 1)
                psum += __shfl_xor(psum, off, 32);
            rsum[r] = rsum[r] * scale[r] + psum;
        }
        // wave-private LDS round-trip (C-layout -> A-layout); DS in-order per wave,
        // the wait is a compile barrier + guarantees the store data has landed.
        asm volatile("s_wait_dscnt 0" ::: "memory");

#pragma unroll
        for (int c = 0; c < 8; ++c)
#pragma unroll
            for (int r = 0; r < 8; ++r) o[c][r] *= scale[r];

        v16h pa0 = loadA(pw,      72, lane);   // P cols (K) 0..31
        v16h pa1 = loadA(pw + 32, 72, lane);   // P cols (K) 32..63
#pragma unroll
        for (int c = 0; c < 8; ++c) {
            v16h b0 = loadB_nk(Gs[cur] + (c * 16) * 72,      72, lane);  // K(m)=0..31
            o[c] = wmma16(pa0, b0, o[c]);
            v16h b1 = loadB_nk(Gs[cur] + (c * 16) * 72 + 32, 72, lane);  // K(m)=32..63
            o[c] = wmma16(pa1, b1, o[c]);
        }

        // next tile must be fully in LDS (and all waves done reading) before swap
        wait_async0();
        __syncthreads();
    }

    // finalize: divide by row sums, store f32
#pragma unroll
    for (int c = 0; c < 8; ++c)
#pragma unroll
        for (int r = 0; r < 8; ++r) {
            int row = rg0 + ((lane >> 4) << 3) + r;
            y[(size_t)row * CI_DIM + c * 16 + (lane & 15)] = o[c][r] / rsum[r];
        }
}

// ---------------- final GEMM: [NPIX,128] x Bt[256,128], *sita + residual ----------------

__global__ void __launch_bounds__(128) final_gemm(const _Float16* __restrict__ A,
                                                  const _Float16* __restrict__ Bt,
                                                  const float* __restrict__ bias,
                                                  const float* __restrict__ resid,
                                                  const float* __restrict__ sita,
                                                  float* __restrict__ out) {
    __shared__ _Float16 As[64 * 136];    // A tile  [m][k=128]
    __shared__ _Float16 Bs[256 * 136];   // weightT [n=256][k=128]
    const int tid  = threadIdx.x;
    const int wave = tid >> 5;
    const int lane = tid & 31;
    const int m0   = blockIdx.x * 64;

    g2s_async(As, 136, A + (size_t)m0 * CI_DIM, CI_DIM, 64, 128, tid, 128);
    g2s_async(Bs, 136, Bt, CI_DIM, 256, 128, tid, 128);
    wait_async0();
    __syncthreads();

    const float sc = sita[0];
    v16h af[4];
#pragma unroll
    for (int kc = 0; kc < 4; ++kc)
        af[kc] = loadA(As + (wave * 16) * 136 + kc * 32, 136, lane);

    const int rbase = m0 + wave * 16 + ((lane >> 4) << 3);
    const int cb    = lane & 15;
#pragma unroll
    for (int n = 0; n < 16; ++n) {
        v8f acc = zero8();
#pragma unroll
        for (int kc = 0; kc < 4; ++kc) {
            v16h b = loadB_nk(Bs + (n * 16) * 136 + kc * 32, 136, lane);
            acc = wmma16(af[kc], b, acc);
        }
        int col  = n * 16 + cb;
        float bb = bias[col];
#pragma unroll
        for (int r = 0; r < 8; ++r) {
            size_t idx = (size_t)(rbase + r) * C_DIM + col;
            out[idx] = (acc[r] + bb) * sc + resid[idx];
        }
    }
}

// ---------------- host launch ----------------

extern "C" void kernel_launch(void* const* d_in, const int* in_sizes, int n_in,
                              void* d_out, int out_size, void* d_ws, size_t ws_size,
                              hipStream_t stream) {
    const float* x_in  = (const float*)d_in[0];
    const float* bn1g  = (const float*)d_in[1];
    const float* bn1b  = (const float*)d_in[2];
    const float* bn1m  = (const float*)d_in[3];
    const float* bn1v  = (const float*)d_in[4];
    const float* bn2g  = (const float*)d_in[5];
    const float* bn2b  = (const float*)d_in[6];
    const float* bn2m  = (const float*)d_in[7];
    const float* bn2v  = (const float*)d_in[8];
    const float* W_g   = (const float*)d_in[9];
    const float* b_g   = (const float*)d_in[10];
    const float* W_th  = (const float*)d_in[11];
    const float* b_th  = (const float*)d_in[12];
    const float* W_ph  = (const float*)d_in[13];
    const float* b_ph  = (const float*)d_in[14];
    const float* W_z   = (const float*)d_in[15];
    const float* b_z   = (const float*)d_in[16];
    const float* sita  = (const float*)d_in[17];

    // workspace carve (halves)
    _Float16* xh  = (_Float16*)d_ws;                       // NPIX*256
    _Float16* th  = xh  + (size_t)NPIX * C_DIM;            // NPIX*128
    _Float16* gf  = th  + (size_t)NPIX * CI_DIM;
    _Float16* pf  = gf  + (size_t)NPIX * CI_DIM;
    _Float16* gpt = pf  + (size_t)NPIX * CI_DIM;           // B*128*NS (channel-major)
    _Float16* pp  = gpt + (size_t)BATCH * NS * CI_DIM;     // B*NS*128
    _Float16* fre = pp  + (size_t)BATCH * NS * CI_DIM;     // NPIX*128
    _Float16* wgh = fre + (size_t)NPIX * CI_DIM;           // transposed weights (f16)
    _Float16* wth = wgh + C_DIM * CI_DIM;
    _Float16* wph = wth + C_DIM * CI_DIM;
    _Float16* wzh = wph + C_DIM * CI_DIM;
    float*    yv  = (float*)(wzh + CI_DIM * C_DIM);        // NPIX*128 f32

    // 1) weight conversion f32 -> f16, transposed to [n][k]
    cvt_transpose<<<128, 256, 0, stream>>>(W_g,  wgh, C_DIM, CI_DIM);
    cvt_transpose<<<128, 256, 0, stream>>>(W_th, wth, C_DIM, CI_DIM);
    cvt_transpose<<<128, 256, 0, stream>>>(W_ph, wph, C_DIM, CI_DIM);
    cvt_transpose<<<128, 256, 0, stream>>>(W_z,  wzh, CI_DIM, C_DIM);

    // 2) x = elu(bn1(inputs)) -> f16
    bn_elu_f16<<<NPIX * C_DIM / 256, 256, 0, stream>>>(
        x_in, bn1g, bn1b, bn1m, bn1v, xh, C_DIM - 1, NPIX * C_DIM);

    // 3) projections (WMMA)
    proj_gemm<<<NPIX / 64, 128, 0, stream>>>(xh, wth, b_th, th);
    proj_gemm<<<NPIX / 64, 128, 0, stream>>>(xh, wgh, b_g,  gf);
    proj_gemm<<<NPIX / 64, 128, 0, stream>>>(xh, wph, b_ph, pf);

    // 4) width-2 max pool (g goes out channel-major)
    pool_w2<<<BATCH * NS * CI_DIM / 256, 256, 0, stream>>>(
        gf, pf, gpt, pp, BATCH * NS * CI_DIM);

    // 5) fused attention (WMMA, online softmax, async double-buffered tiles) -> y f32
    attn_flash<<<BATCH * (NQ / 64), 128, 0, stream>>>(th, pp, gpt, yv);

    // 6) fre = elu(bn2(y)) -> f16
    bn_elu_f16<<<NPIX * CI_DIM / 256, 256, 0, stream>>>(
        yv, bn2g, bn2b, bn2m, bn2v, fre, CI_DIM - 1, NPIX * CI_DIM);

    // 7) out = (fre @ W_z + b_z) * sita + inputs  (WMMA)
    final_gemm<<<NPIX / 64, 128, 0, stream>>>(
        fre, wzh, b_z, x_in, sita, (float*)d_out);
}
